// PuzzleMoE_68667937129030
// MI455X (gfx1250) — matrix operand order repaired and checked by reading, not compiled, not run on hardware
//
#include <hip/hip_runtime.h>
#include <cstdint>
#include <cstddef>

// ---------- CDNA5 WMMA types ----------
typedef __attribute__((ext_vector_type(16))) __bf16 v16bf;
typedef __attribute__((ext_vector_type(8)))  float  v8f;
typedef __attribute__((ext_vector_type(8)))  unsigned short ush8;

union FragU { v16bf v; ush8 h[2]; unsigned short s[16]; };

// hardware f32 -> bf16 (RNE) via v_cvt
__device__ __forceinline__ unsigned short f2bf_bits(float f) {
  union { __bf16 b; unsigned short s; } u;
  u.b = (__bf16)f;
  return u.s;
}
__device__ __forceinline__ float bf2f(unsigned short h) {
  union { unsigned u; float f; } v; v.u = ((unsigned)h) << 16;
  return v.f;
}
__device__ __forceinline__ v8f wmma_bf16(v16bf a, v16bf b, v8f c) {
  // (neg_a, A, neg_b, B, c_mod, C, reuse_a, reuse_b)
  return __builtin_amdgcn_wmma_f32_16x16x32_bf16(false, a, false, b, (short)0, c,
                                                 false, false);
}
__device__ __forceinline__ v8f vzero8() {
  v8f z;
#pragma unroll
  for (int i = 0; i < 8; ++i) z[i] = 0.0f;
  return z;
}

// A fragment (16x32 bf16) from row-major LDS; row = &A[m][k0]; hlf = lane>>4.
// ISA layout: lane m holds K chunks [hlf*8, +8) and [16+hlf*8, +8).
__device__ __forceinline__ v16bf a_frag_rm(const unsigned short* row, int hlf) {
  FragU f;
  f.h[0] = *(const ush8*)(row + hlf * 8);
  f.h[1] = *(const ush8*)(row + 16 + hlf * 8);
  return f.v;
}
// A fragment for the implicit-im2col conv view: K = tap*ldc + c, A[t][k] = X[(t+tap)*ldc + c].
// 8-aligned K chunks never cross a tap boundary -> each chunk is one aligned ds_load_b128.
__device__ __forceinline__ v16bf a_frag_conv(const unsigned short* X, int ldc,
                                             int t, int k0, int hlf) {
  int kb0 = k0 + hlf * 8;
  int kb1 = k0 + 16 + hlf * 8;
  FragU f;
  f.h[0] = *(const ush8*)(X + (t + kb0 / ldc) * ldc + (kb0 % ldc));
  f.h[1] = *(const ush8*)(X + (t + kb1 / ldc) * ldc + (kb1 % ldc));
  return f.v;
}
// B fragment (32x16 bf16) from LDS row-major W[k][n]; p0 = &W[k0][n0]; lane%16 = n, K chunks [hlf*8,+8) and [16+hlf*8,+8).
__device__ __forceinline__ v16bf b_frag_lds(const unsigned short* p0, int ldn, int lane) {
  const unsigned short* p = p0 + (lane & 15);
  const int hb = (lane >> 4) * 8;
  FragU f;
#pragma unroll
  for (int s = 0; s < 16; ++s) {
    const int k = (s < 8) ? (hb + s) : (16 + hb + s - 8);
    f.s[s] = p[k * ldn];
  }
  return f.v;
}
// B fragment from pre-converted bf16 weights in global memory (16 strided 16-bit loads, one per k).
__device__ __forceinline__ v16bf b_frag_gbf16(const unsigned short* __restrict__ W,
                                              int ldn, int k0, int n0, int lane) {
  const unsigned short* p = W + (size_t)k0 * ldn + n0 + (lane & 15);
  const int hb = (lane >> 4) * 8;
  FragU f;
#pragma unroll
  for (int s = 0; s < 16; ++s) {
    const int k = (s < 8) ? (hb + s) : (16 + hb + s - 8);
    f.s[s] = p[(size_t)k * ldn];
  }
  return f.v;
}

// ---------------- problem constants ----------------
#define NB   512
#define NP   9
#define NTOK 4608      // NB*NP
#define NC_  12
#define LL   256
#define HH   64
#define DD   512
#define EE   8
#define NCLS 128

// ================= K0: zero router counters =================
__global__ void k_zero_cnt(int* cnt) {
  if (threadIdx.x < EE) cnt[threadIdx.x] = 0;
}

// ================= Kprep: one-shot weight convert/swizzle (bf16) =================
// conv weights go to the WMMA-B [k][n] layout (zero-padded); big weights plain bf16.
__global__ void __launch_bounds__(256)
k_prep_weights(const float* __restrict__ c1w, const float* __restrict__ c2w,
               const float* __restrict__ proj_w,
               const float* __restrict__ exp_w1, const float* __restrict__ exp_w2,
               unsigned short* __restrict__ w1s, unsigned short* __restrict__ w2s,
               unsigned short* __restrict__ projbf,
               unsigned short* __restrict__ w1bf, unsigned short* __restrict__ w2bf) {
  const int stride = gridDim.x * 256;
  const int t0 = blockIdx.x * 256 + threadIdx.x;
  // conv1: [k = tap*16 + c][n], 96x64, pad tap>=5 / c>=12 with zero
  for (int i = t0; i < 96 * 64; i += stride) {
    int k = i >> 6, n = i & 63, tap = k >> 4, c = k & 15;
    float v = (tap < 5 && c < 12) ? c1w[n * 60 + c * 5 + tap] : 0.0f;
    w1s[i] = f2bf_bits(v);
  }
  // conv2: [k = tap*64 + ch][n], 192x64
  for (int i = t0; i < 192 * 64; i += stride) {
    int k = i >> 6, n = i & 63, tap = k >> 6, ch = k & 63;
    w2s[i] = f2bf_bits(c2w[n * 192 + ch * 3 + tap]);
  }
  for (int i = t0; i < HH * DD; i += stride) projbf[i] = f2bf_bits(proj_w[i]);
  for (size_t i = t0; i < (size_t)EE * DD * DD; i += stride) w1bf[i] = f2bf_bits(exp_w1[i]);
  for (size_t i = t0; i < (size_t)EE * DD * NCLS; i += stride) w2bf[i] = f2bf_bits(exp_w2[i]);
}

// ================= K1: conv1+conv2+meanpool -> feat(bf16) =================
// one block per patch; 8 waves. conv1: (256 x 96pad)x(96x64); conv2: (256 x 192)x(192x64)
__global__ void __launch_bounds__(256)
k_conv_feat(const float* __restrict__ patches,
            const unsigned short* __restrict__ w1s, const float* __restrict__ c1b,
            const unsigned short* __restrict__ w2s, const float* __restrict__ c2b,
            unsigned short* __restrict__ featbf) {
  extern __shared__ char smem[];
  unsigned short* xin = (unsigned short*)smem;     // 264 x 16   (pos q stored at row q+2)
  unsigned short* w1  = xin + 264 * 16;            // 96  x 64
  unsigned short* w2  = w1 + 96 * 64;              // 192 x 64
  unsigned short* h1b = w2 + 192 * 64;             // 258 x 64   (pos p stored at row p+1)
  float* b1s  = (float*)(h1b + 258 * 64);          // 64
  float* b2s  = b1s + 64;                          // 64
  float* fsum = b2s + 64;                          // 64

  const int tid  = threadIdx.x;
  const int lane = tid & 31, wave = tid >> 5;
  const int hlf  = lane >> 4, c15 = lane & 15;
  const int patch = blockIdx.x;

  // zero the regions whose padding/halo is consumed by WMMA fragments
  for (int i = tid; i < 264 * 16; i += 256) xin[i] = 0;
  for (int i = tid; i < 258 * 64; i += 256) h1b[i] = 0;
  if (tid < 64) fsum[tid] = 0.0f;
  __syncthreads();

  // fill staging: input converts f32->bf16 (hardware cvt); weights are plain copies
  const float* xg = patches + (size_t)patch * NC_ * LL;
  for (int i = tid; i < NC_ * LL; i += 256) {
    int c = i >> 8, q = i & 255;
    xin[(q + 2) * 16 + c] = f2bf_bits(xg[i]);
  }
  for (int i = tid; i < 96 * 64; i += 256) w1[i] = w1s[i];
  for (int i = tid; i < 192 * 64; i += 256) w2[i] = w2s[i];
  if (tid < 64) { b1s[tid] = c1b[tid]; b2s[tid] = c2b[tid]; }
  __syncthreads();

  // ---- conv1 via WMMA ----
  for (int mi = 0; mi < 2; ++mi) {
    const int t0 = (wave * 2 + mi) * 16;
    v8f acc[4] = {vzero8(), vzero8(), vzero8(), vzero8()};
#pragma unroll
    for (int ks = 0; ks < 3; ++ks) {
      const int k0 = ks * 32;
      v16bf a = a_frag_conv(xin, 16, t0 + c15, k0, hlf);
#pragma unroll
      for (int nt = 0; nt < 4; ++nt)
        acc[nt] = wmma_bf16(a, b_frag_lds(w1 + k0 * 64 + nt * 16, 64, lane), acc[nt]);
    }
#pragma unroll
    for (int nt = 0; nt < 4; ++nt) {
      const int n = nt * 16 + c15;
      const float bias = b1s[n];
#pragma unroll
      for (int r = 0; r < 8; ++r) {
        const int t = t0 + r + 8 * hlf;
        float v = acc[nt][r] + bias;
        h1b[(t + 1) * 64 + n] = f2bf_bits(v > 0.0f ? v : 0.0f);
      }
    }
  }
  __syncthreads();

  // ---- conv2 via WMMA + relu + mean pool ----
  for (int mi = 0; mi < 2; ++mi) {
    const int t0 = (wave * 2 + mi) * 16;
    v8f acc[4] = {vzero8(), vzero8(), vzero8(), vzero8()};
#pragma unroll 2
    for (int ks = 0; ks < 6; ++ks) {
      const int k0 = ks * 32;
      v16bf a = a_frag_conv(h1b, 64, t0 + c15, k0, hlf);
#pragma unroll
      for (int nt = 0; nt < 4; ++nt)
        acc[nt] = wmma_bf16(a, b_frag_lds(w2 + k0 * 64 + nt * 16, 64, lane), acc[nt]);
    }
#pragma unroll
    for (int nt = 0; nt < 4; ++nt) {
      const int n = nt * 16 + c15;
      const float bias = b2s[n];
      float s = 0.0f;
#pragma unroll
      for (int r = 0; r < 8; ++r) {
        float v = acc[nt][r] + bias;
        s += (v > 0.0f ? v : 0.0f);
      }
      atomicAdd(&fsum[n], s);                      // ds_add_f32
    }
  }
  __syncthreads();
  if (tid < 64) featbf[(size_t)patch * HH + tid] = f2bf_bits(fsum[tid] * (1.0f / 256.0f));
}

// ================= K2: proj + pos + gate + top-2 router =================
// one block per 16 tokens (288 blocks)
__global__ void __launch_bounds__(256)
k_proj_gate(const unsigned short* __restrict__ featbf,
            const unsigned short* __restrict__ projbf, const float* __restrict__ proj_b,
            const float* __restrict__ pos,
            const float* __restrict__ gate_w, const float* __restrict__ gate_b,
            unsigned short* __restrict__ embbf,
            int* __restrict__ cnt, int* __restrict__ a_tok, float* __restrict__ a_wt) {
  extern __shared__ char smem[];
  unsigned short* featA = (unsigned short*)smem;   // 16 x 64
  unsigned short* embb  = featA + 16 * 64;         // 16 x 512
  float* gw = (float*)(embb + 16 * 512);           // 512 x 8
  float* lg = gw + 512 * 8;                        // 16 x 8

  const int tid  = threadIdx.x;
  const int lane = tid & 31, wave = tid >> 5;
  const int hlf  = lane >> 4, c15 = lane & 15;
  const int mt   = blockIdx.x;

  for (int i = tid; i < 16 * 64; i += 256) featA[i] = featbf[(size_t)mt * 16 * 64 + i];
  for (int i = tid; i < 512 * 8; i += 256) gw[i] = gate_w[i];
  __syncthreads();

  // emb tile: (16 x 64) @ (64 x 512)
  v8f acc[4] = {vzero8(), vzero8(), vzero8(), vzero8()};
#pragma unroll
  for (int ks = 0; ks < 2; ++ks) {
    const int k0 = ks * 32;
    v16bf a = a_frag_rm(featA + c15 * 64 + k0, hlf);
#pragma unroll
    for (int nt = 0; nt < 4; ++nt)
      acc[nt] = wmma_bf16(a, b_frag_gbf16(projbf, DD, k0, wave * 64 + nt * 16, lane), acc[nt]);
  }
#pragma unroll
  for (int nt = 0; nt < 4; ++nt) {
    const int n = wave * 64 + nt * 16 + c15;
    const float pb = proj_b[n];
#pragma unroll
    for (int r = 0; r < 8; ++r) {
      const int m = r + 8 * hlf;
      const int token = mt * 16 + m;
      const int p = token % NP;
      float v = acc[nt][r] + pb + pos[p * DD + n];
      unsigned short hb = f2bf_bits(v);
      embb[m * DD + n] = hb;
      embbf[(size_t)token * DD + n] = hb;
    }
  }
  __syncthreads();

  // gate logits (tiny: 16x512x8)
  if (tid < 128) {
    const int m = tid >> 3, e = tid & 7;
    float s = gate_b[e];
    for (int n = 0; n < DD; ++n) s += bf2f(embb[m * DD + n]) * gw[n * 8 + e];
    lg[m * 8 + e] = s;
  }
  __syncthreads();

  // top-2 softmax router; push (token*2+slot, w/9) to per-expert lists
  if (tid < 16) {
    const int m = tid, token = mt * 16 + m;
    float best = -3.4e38f; int bi = 0;
    for (int e = 0; e < EE; ++e) { float v = lg[m * 8 + e]; if (v > best) { best = v; bi = e; } }
    float sec = -3.4e38f; int si = (bi == 0) ? 1 : 0;
    for (int e = 0; e < EE; ++e) {
      if (e == bi) continue;
      float v = lg[m * 8 + e]; if (v > sec) { sec = v; si = e; }
    }
    const float r  = __expf(sec - best);
    const float w0 = 1.0f / (1.0f + r);
    const float w1 = r / (1.0f + r);
    int s0 = atomicAdd(&cnt[bi], 1);
    a_tok[bi * NTOK + s0] = token * 2;
    a_wt [bi * NTOK + s0] = w0 * (1.0f / 9.0f);
    int s1 = atomicAdd(&cnt[si], 1);
    a_tok[si * NTOK + s1] = token * 2 + 1;
    a_wt [si * NTOK + s1] = w1 * (1.0f / 9.0f);
  }
}

// ================= K3: sparse expert FFN (top-2 only) =================
// grid (288, 8): expert e, 16-assignment tile; early-exit beyond count.
__global__ void __launch_bounds__(256)
k_expert(const unsigned short* __restrict__ embbf,
         const unsigned short* __restrict__ w1bf, const float* __restrict__ exp_b1,
         const unsigned short* __restrict__ w2bf, const float* __restrict__ exp_b2,
         const int* __restrict__ cnt, const int* __restrict__ a_tok,
         const float* __restrict__ a_wt, float* __restrict__ contrib) {
  const int e = blockIdx.y;
  const int base = blockIdx.x * 16;
  const int count = cnt[e];
  if (base >= count) return;

  extern __shared__ char smem[];
  int*   tok = (int*)smem;                              // 16
  float* wts = (float*)(smem + 64);                     // 16
  unsigned short* embA = (unsigned short*)(smem + 256); // 16 x 512
  unsigned short* h1s  = embA + 16 * 512;               // 16 x 512
  float* b1s = (float*)(h1s + 16 * 512);                // 512

  const int tid  = threadIdx.x;
  const int lane = tid & 31, wave = tid >> 5;
  const int hlf  = lane >> 4, c15 = lane & 15;

  if (tid < 16) {
    const int i = base + tid;
    if (i < count) { tok[tid] = a_tok[e * NTOK + i]; wts[tid] = a_wt[e * NTOK + i]; }
    else           { tok[tid] = 0;                   wts[tid] = 0.0f; }
  }
  for (int i = tid; i < DD; i += 256) b1s[i] = exp_b1[e * DD + i];
  __syncthreads();

  { // gather routed emb rows (bf16) into LDS
    const int m = tid >> 4, seg = tid & 15;
    const ush8* src = (const ush8*)(embbf + (size_t)(tok[m] >> 1) * DD + seg * 32);
    ush8* dst = (ush8*)(embA + m * DD + seg * 32);
    dst[0] = src[0]; dst[1] = src[1]; dst[2] = src[2]; dst[3] = src[3];
  }
  __syncthreads();

  // h1 = relu(A @ W1_e + b1): (16 x 512) x (512 x 512), 4 N-tiles per wave
  const unsigned short* W1 = w1bf + (size_t)e * DD * DD;
  v8f acc[4] = {vzero8(), vzero8(), vzero8(), vzero8()};
#pragma unroll 2
  for (int ks = 0; ks < 16; ++ks) {
    const int k0 = ks * 32;
    v16bf a = a_frag_rm(embA + c15 * DD + k0, hlf);
    if (k0 + 32 < DD)  // stream-ahead the next K-step of W1 (global_prefetch_b8)
      __builtin_prefetch(W1 + (size_t)(k0 + 32 + lane) * DD + wave * 64, 0, 1);
#pragma unroll
    for (int nt = 0; nt < 4; ++nt)
      acc[nt] = wmma_bf16(a, b_frag_gbf16(W1, DD, k0, wave * 64 + nt * 16, lane), acc[nt]);
  }
#pragma unroll
  for (int nt = 0; nt < 4; ++nt) {
    const int n = wave * 64 + nt * 16 + c15;
    const float bs = b1s[n];
#pragma unroll
    for (int r = 0; r < 8; ++r) {
      float v = acc[nt][r] + bs;
      h1s[(r + 8 * hlf) * DD + n] = f2bf_bits(v > 0.0f ? v : 0.0f);
    }
  }
  __syncthreads();

  // oe = h1 @ W2_e + b2, one 16-col N-tile per wave; scatter weighted contributions
  const unsigned short* W2 = w2bf + (size_t)e * DD * NCLS;
  v8f o = vzero8();
  const int n0 = wave * 16;
#pragma unroll 4
  for (int ks = 0; ks < 16; ++ks) {
    const int k0 = ks * 32;
    v16bf a = a_frag_rm(h1s + c15 * DD + k0, hlf);
    o = wmma_bf16(a, b_frag_gbf16(W2, NCLS, k0, n0, lane), o);
  }
  const int n = n0 + c15;
  const float b2v = exp_b2[e * NCLS + n];
  int nv = count - base; if (nv > 16) nv = 16;
#pragma unroll
  for (int r = 0; r < 8; ++r) {
    const int m = r + 8 * hlf;
    if (m < nv)
      contrib[(size_t)tok[m] * NCLS + n] = (o[r] + b2v) * wts[m];
  }
}

// ================= K4: reduce 18 contribution rows per batch =================
__global__ void __launch_bounds__(128)
k_reduce_out(const float* __restrict__ contrib, float* __restrict__ out) {
  const int b = blockIdx.x, n = threadIdx.x;
  const float* p = contrib + (size_t)b * 18 * NCLS + n;
  float s = 0.0f;
#pragma unroll
  for (int i = 0; i < 18; ++i) s += p[i * NCLS];
  out[b * NCLS + n] = s;
}

// ================= host launch =================
extern "C" void kernel_launch(void* const* d_in, const int* in_sizes, int n_in,
                              void* d_out, int out_size, void* d_ws, size_t ws_size,
                              hipStream_t stream) {
  (void)in_sizes; (void)n_in; (void)out_size; (void)ws_size;
  const float* patches = (const float*)d_in[0];
  const float* c1w     = (const float*)d_in[1];
  const float* c1b     = (const float*)d_in[2];
  const float* c2w     = (const float*)d_in[3];
  const float* c2b     = (const float*)d_in[4];
  const float* proj_w  = (const float*)d_in[5];
  const float* proj_b  = (const float*)d_in[6];
  const float* pos     = (const float*)d_in[7];
  const float* gate_w  = (const float*)d_in[8];
  const float* gate_b  = (const float*)d_in[9];
  const float* exp_w1  = (const float*)d_in[10];
  const float* exp_b1  = (const float*)d_in[11];
  const float* exp_w2  = (const float*)d_in[12];
  const float* exp_b2  = (const float*)d_in[13];

  // workspace layout (bytes), 256-aligned; total ~15.7 MB
  char* ws = (char*)d_ws;
  int* cnt               = (int*)(ws + 0);                       // 8
  int* a_tok             = (int*)(ws + 256);                     // 8*4608 i32
  float* a_wt            = (float*)(ws + 147712);                // 8*4608 f32
  unsigned short* featbf = (unsigned short*)(ws + 295168);       // 4608*64 bf16
  unsigned short* embbf  = (unsigned short*)(ws + 884992);       // 4608*512 bf16
  float* contrib         = (float*)(ws + 5603584);               // 9216*128 f32
  unsigned short* w1s    = (unsigned short*)(ws + 10322176);     // 96*64 bf16 (conv1 B)
  unsigned short* w2s    = (unsigned short*)(ws + 10334464);     // 192*64 bf16 (conv2 B)
  unsigned short* projbf = (unsigned short*)(ws + 10359040);     // 64*512 bf16
  unsigned short* w1bf   = (unsigned short*)(ws + 10424576);     // 8*512*512 bf16
  unsigned short* w2bf   = (unsigned short*)(ws + 14618880);     // 8*512*128 bf16

  k_zero_cnt<<<1, 64, 0, stream>>>(cnt);
  k_prep_weights<<<512, 256, 0, stream>>>(c1w, c2w, proj_w, exp_w1, exp_w2,
                                          w1s, w2s, projbf, w1bf, w2bf);

  // conv LDS: xin 8448 + w1 12288 + w2 24576 + h1b 33024 + 3*256 = 79104 B
  k_conv_feat<<<4608, 256, 79104, stream>>>(patches, w1s, c1b, w2s, c2b, featbf);

  // proj LDS: 2048 + 16384 + 16384 + 512 = 35328 B
  k_proj_gate<<<288, 256, 35328, stream>>>(featbf, projbf, proj_b, pos,
                                           gate_w, gate_b, embbf, cnt, a_tok, a_wt);

  // expert LDS: 256 + 16384 + 16384 + 2048 = 35072 B
  k_expert<<<dim3(288, 8), 256, 35072, stream>>>(embbf, w1bf, exp_b1, w2bf, exp_b2,
                                                 cnt, a_tok, a_wt, contrib);

  k_reduce_out<<<512, 128, 0, stream>>>(contrib, (float*)d_out);
}
